// Graph_CNN_Mesh_Pose_10015863734924
// MI455X (gfx1250) — matrix operand order repaired and checked by reading, not compiled, not run on hardware
//
#include <hip/hip_runtime.h>
#include <hip/hip_bf16.h>

// ---------------------------------------------------------------------------
// Problem constants (from the reference)
// ---------------------------------------------------------------------------
#define BB   512
#define V0   1024
#define V1   256
#define V2   64
#define E0   8192
#define E1   2048

typedef __attribute__((ext_vector_type(16))) __bf16          v16bf;
typedef __attribute__((ext_vector_type(16))) unsigned short  v16u;
typedef __attribute__((ext_vector_type(8)))  unsigned int    v8u;
typedef __attribute__((ext_vector_type(8)))  float           v8f;

// pack two floats into a dword of two bf16 (RNE via native __bf16 convert)
__device__ __forceinline__ unsigned pk2(float a, float b) {
    unsigned short x = __builtin_bit_cast(unsigned short, (__bf16)a);
    unsigned short y = __builtin_bit_cast(unsigned short, (__bf16)b);
    return (unsigned)x | ((unsigned)y << 16);
}
__device__ __forceinline__ unsigned short bf1(float a) {
    return __builtin_bit_cast(unsigned short, (__bf16)a);
}

// ---------------------------------------------------------------------------
// CDNA5 async global->LDS bulk copy (ASYNCcnt-tracked DMA-style path).
// Each thread moves 16-byte chunks; wave waits its own ASYNCcnt, caller
// must __syncthreads() afterwards so all waves' data is visible.
// ---------------------------------------------------------------------------
__device__ __forceinline__ void async_copy_to_lds(const float* __restrict__ gsrc,
                                                  float* lds_dst, int nfloats) {
    int nchunks = nfloats >> 2;                      // 16B granules
    for (int c = threadIdx.x; c < nchunks; c += blockDim.x) {
        unsigned dst = (unsigned)(size_t)(lds_dst + c * 4);      // LDS byte offset
        unsigned long long src = (unsigned long long)(gsrc + c * 4);
        asm volatile("global_load_async_to_lds_b128 %0, %1, off"
                     :: "v"(dst), "v"(src) : "memory");
    }
    asm volatile("s_wait_asynccnt 0" ::: "memory");
}

// ---------------------------------------------------------------------------
// Layout transform:  x[B,V0,3] -> X0[V0,B,3]
// ---------------------------------------------------------------------------
__global__ void k_transpose_in(const float* __restrict__ x, float* __restrict__ X0) {
    int idx = blockIdx.x * blockDim.x + threadIdx.x;     // over V0*B*3
    if (idx >= V0 * BB * 3) return;
    int c = idx % 3;
    int t = idx / 3;
    int b = t % BB;
    int v = t / BB;
    X0[idx] = x[((size_t)b * V0 + v) * 3 + c];
}

// ---------------------------------------------------------------------------
// t1 = 0 ; t2 = -x0   (prepares x2 = 2*L*x1 - x0 accumulation), float4-wide
// ---------------------------------------------------------------------------
__global__ void k_init_t4(const float* __restrict__ x0, float* __restrict__ t1,
                          float* __restrict__ t2, int n4) {
    int i = blockIdx.x * blockDim.x + threadIdx.x;
    if (i >= n4) return;
    float4 z = {0.f, 0.f, 0.f, 0.f};
    float4 v = ((const float4*)x0)[i];
    ((float4*)t1)[i] = z;
    float4 nv = {-v.x, -v.y, -v.z, -v.w};
    ((float4*)t2)[i] = nv;
}

// ---------------------------------------------------------------------------
// COO SpMM scatter (float4 granule):
//   Y[rows[e], :, :] += alpha * vals[e] * X[cols[e], :, :]
// ---------------------------------------------------------------------------
__global__ void k_spmm_scatter4(const int* __restrict__ rows, const int* __restrict__ cols,
                                const float* __restrict__ vals, const float* __restrict__ X,
                                float* __restrict__ Y, float alpha, int E, int BF) {
    int BF4 = BF >> 2;
    long long total = (long long)E * BF4;
    for (long long idx = (long long)blockIdx.x * blockDim.x + threadIdx.x; idx < total;
         idx += (long long)gridDim.x * blockDim.x) {
        int e    = (int)(idx / BF4);
        int rem  = (int)(idx % BF4) << 2;
        float s  = alpha * vals[e];
        float4 xv = *(const float4*)(X + (size_t)cols[e] * BF + rem);
        float* yp = Y + (size_t)rows[e] * BF + rem;
        __hip_atomic_fetch_add(yp + 0, s * xv.x, __ATOMIC_RELAXED, __HIP_MEMORY_SCOPE_AGENT);
        __hip_atomic_fetch_add(yp + 1, s * xv.y, __ATOMIC_RELAXED, __HIP_MEMORY_SCOPE_AGENT);
        __hip_atomic_fetch_add(yp + 2, s * xv.z, __ATOMIC_RELAXED, __HIP_MEMORY_SCOPE_AGENT);
        __hip_atomic_fetch_add(yp + 3, s * xv.w, __ATOMIC_RELAXED, __HIP_MEMORY_SCOPE_AGENT);
    }
}

// ---------------------------------------------------------------------------
// Scalar-gather Chebyshev GEMM (layer 0 only: KD = 9 padded to 32)
// contraction order i = fin*3 + k  (reference order)
// ---------------------------------------------------------------------------
template <int FIN, int FOUT>
__global__ void k_cheby_gemm_s(const float* __restrict__ x0, const float* __restrict__ x1,
                               const float* __restrict__ x2, const float* __restrict__ W,
                               const float* __restrict__ bias, float* __restrict__ out,
                               int Mrows) {
    constexpr int KD = FIN * 3;
    constexpr int KP = (KD + 31) & ~31;
    constexpr int NT = FOUT / 16;
    constexpr int KT = KP / 32;

    __shared__ unsigned int wlds[FOUT * KP / 2];
    for (int idx = threadIdx.x; idx < FOUT * KP / 2; idx += blockDim.x) {
        int o  = idx / (KP / 2);
        int ip = (idx % (KP / 2)) * 2;
        float lo = (ip     < KD) ? W[(size_t)o * KD + ip]     : 0.0f;
        float hi = (ip + 1 < KD) ? W[(size_t)o * KD + ip + 1] : 0.0f;
        wlds[idx] = pk2(lo, hi);
    }
    __syncthreads();

    int wave = blockIdx.x * (blockDim.x / 32) + (threadIdx.x / 32);
    int lane = threadIdx.x & 31;
    int r0   = wave * 16;
    if (r0 >= Mrows) return;
    int m = lane & 15, h = lane >> 4;
    int row = r0 + m;

    const v8f zacc = {0.f, 0.f, 0.f, 0.f, 0.f, 0.f, 0.f, 0.f};
    v8f acc[NT];
#pragma unroll
    for (int t = 0; t < NT; ++t) acc[t] = zacc;

    for (int kk = 0; kk < KT; ++kk) {
        v16u au;
#pragma unroll
        for (int j = 0; j < 8; ++j) {
            int kbase = 16 * h + 2 * j;
            int i0 = kk * 32 + kbase;
            float e0 = 0.0f, e1 = 0.0f;
            if (i0 < KD) {
                int fin = i0 / 3, c = i0 % 3;
                const float* p = (c == 0) ? x0 : ((c == 1) ? x1 : x2);
                e0 = p[(size_t)row * FIN + fin];
            }
            if (i0 + 1 < KD) {
                int fin = (i0 + 1) / 3, c = (i0 + 1) % 3;
                const float* p = (c == 0) ? x0 : ((c == 1) ? x1 : x2);
                e1 = p[(size_t)row * FIN + fin];
            }
            au[2 * j]     = bf1(e0);
            au[2 * j + 1] = bf1(e1);
        }
        v16bf a = __builtin_bit_cast(v16bf, au);

#pragma unroll
        for (int t = 0; t < NT; ++t) {
            v8u bu;
            int o = t * 16 + m;
#pragma unroll
            for (int j = 0; j < 8; ++j)
                bu[j] = wlds[(size_t)o * (KP / 2) + kk * 16 + 8 * h + j];
            v16bf b = __builtin_bit_cast(v16bf, bu);
            acc[t] = __builtin_amdgcn_wmma_f32_16x16x32_bf16(
                false, a, false, b, (short)0, acc[t], false, false);
        }
    }
#pragma unroll
    for (int t = 0; t < NT; ++t) {
        int o = t * 16 + m;
        float bv = bias[o];
#pragma unroll
        for (int rr = 0; rr < 8; ++rr)
            out[(size_t)(r0 + h * 8 + rr) * FOUT + o] = acc[t][rr] + bv;
    }
}

// ---------------------------------------------------------------------------
// Vectorized Chebyshev GEMM (FIN % 32 == 0): contraction permuted k-major,
//   i' = k*FIN + fin ; W async-DMA'd to LDS (f32), then permuted/packed to
//   bf16 pairs in LDS once per block.
// Every per-lane 8-element K-run lies in a single x0/x1/x2 buffer ->
//   A fragment = 4x float4 loads + packed bf16 converts,
//   B fragment = 2x b128 LDS loads.
// ---------------------------------------------------------------------------
template <int FIN, int FOUT>
__global__ void k_cheby_gemm_v(const float* __restrict__ x0, const float* __restrict__ x1,
                               const float* __restrict__ x2, const float* __restrict__ W,
                               const float* __restrict__ bias, float* __restrict__ out,
                               int Mrows) {
    constexpr int KD = FIN * 3;        // divisible by 32
    constexpr int NT = FOUT / 16;
    constexpr int KT = KD / 32;

    __shared__ float        wf32[FOUT * KD];       // raw W image (async DMA)
    __shared__ unsigned int wlds[FOUT * KD / 2];   // k-major packed bf16

    // stage W via CDNA5 async global->LDS copy
    async_copy_to_lds(W, wf32, FOUT * KD);
    __syncthreads();

    // permute to k-major i' = k*FIN + fin and pack to bf16 pairs
    for (int idx = threadIdx.x; idx < FOUT * KD / 2; idx += blockDim.x) {
        int o  = idx / (KD / 2);
        int ip = (idx % (KD / 2)) * 2;          // i' (even)
        int k0 = ip / FIN,       f0 = ip % FIN;
        int k1 = (ip + 1) / FIN, f1 = (ip + 1) % FIN;
        wlds[idx] = pk2(wf32[o * KD + f0 * 3 + k0],
                        wf32[o * KD + f1 * 3 + k1]);
    }
    __syncthreads();

    int wave = blockIdx.x * (blockDim.x / 32) + (threadIdx.x / 32);
    int lane = threadIdx.x & 31;
    int r0   = wave * 16;
    if (r0 >= Mrows) return;
    int m = lane & 15, h = lane >> 4;
    int row = r0 + m;

    const float* bufs[3] = {x0, x1, x2};

    const v8f zacc = {0.f, 0.f, 0.f, 0.f, 0.f, 0.f, 0.f, 0.f};
    v8f acc[NT];
#pragma unroll
    for (int t = 0; t < NT; ++t) acc[t] = zacc;

    for (int kk = 0; kk < KT; ++kk) {
        v8u au;
        // run 1: K = kk*32 + 16h .. +7 ; run 2: +8
#pragma unroll
        for (int rset = 0; rset < 2; ++rset) {
            int s  = kk * 32 + 16 * h + 8 * rset;   // i' start, 8-aligned
            int kc = s / FIN;                        // which Chebyshev component
            int f0 = s % FIN;                        // run stays inside one buffer
            const float* p = bufs[kc] + (size_t)row * FIN + f0;
            float4 u = *(const float4*)(p);
            float4 v = *(const float4*)(p + 4);
            au[4 * rset + 0] = pk2(u.x, u.y);
            au[4 * rset + 1] = pk2(u.z, u.w);
            au[4 * rset + 2] = pk2(v.x, v.y);
            au[4 * rset + 3] = pk2(v.z, v.w);
        }
        v16bf a = __builtin_bit_cast(v16bf, au);

#pragma unroll
        for (int t = 0; t < NT; ++t) {
            int o = t * 16 + m;
            const uint4* wp = (const uint4*)(wlds + (size_t)o * (KD / 2) + kk * 16 + 8 * h);
            uint4 b0 = wp[0];
            uint4 b1 = wp[1];
            v8u bu;
            bu[0] = b0.x; bu[1] = b0.y; bu[2] = b0.z; bu[3] = b0.w;
            bu[4] = b1.x; bu[5] = b1.y; bu[6] = b1.z; bu[7] = b1.w;
            v16bf b = __builtin_bit_cast(v16bf, bu);
            acc[t] = __builtin_amdgcn_wmma_f32_16x16x32_bf16(
                false, a, false, b, (short)0, acc[t], false, false);
        }
    }
#pragma unroll
    for (int t = 0; t < NT; ++t) {
        int o = t * 16 + m;
        float bv = bias[o];
#pragma unroll
        for (int rr = 0; rr < 8; ++rr)
            out[(size_t)(r0 + h * 8 + rr) * FOUT + o] = acc[t][rr] + bv;
    }
}

// ---------------------------------------------------------------------------
// MaxPool over vertex axis, groups of 4:  in[Vout*4, B, F] -> out[Vout, B, F]
// ---------------------------------------------------------------------------
__global__ void k_pool4(const float* __restrict__ in, float* __restrict__ out,
                        int Vout, int F) {
    int idx = blockIdx.x * blockDim.x + threadIdx.x;
    int total = Vout * BB * F;
    if (idx >= total) return;
    int f  = idx % F;
    int t  = idx / F;
    int b  = t % BB;
    int vp = t / BB;
    size_t base   = ((size_t)(4 * vp) * BB + b) * F + f;
    size_t stride = (size_t)BB * F;
    float m0 = in[base];
    m0 = fmaxf(m0, in[base + stride]);
    m0 = fmaxf(m0, in[base + 2 * stride]);
    m0 = fmaxf(m0, in[base + 3 * stride]);
    out[idx] = m0;
}

// ---------------------------------------------------------------------------
// f32 -> bf16 bulk convert, 8 elements/thread (b128 in, b128 out)
// ---------------------------------------------------------------------------
__global__ void k_cvt_bf16x8(const float* __restrict__ src, unsigned short* __restrict__ dst,
                             int n8) {
    int i = blockIdx.x * blockDim.x + threadIdx.x;
    if (i >= n8) return;
    size_t base = (size_t)i * 8;
    float4 a = *(const float4*)(src + base);
    float4 b = *(const float4*)(src + base + 4);
    uint4 o;
    o.x = pk2(a.x, a.y); o.y = pk2(a.z, a.w);
    o.z = pk2(b.x, b.y); o.w = pk2(b.z, b.w);
    *(uint4*)(dst + base) = o;
}

// fcW2 [63,512] -> bf16 [64,512] with row 63 zero-padded (kills N-edge guards)
__global__ void k_cvt_w2pad(const float* __restrict__ src, unsigned short* __restrict__ dst) {
    int idx = blockIdx.x * blockDim.x + threadIdx.x;   // over 64*512
    if (idx >= 64 * 512) return;
    int o = idx / 512;
    dst[idx] = (o < 63) ? bf1(src[idx]) : (unsigned short)0;
}

// ---------------------------------------------------------------------------
// FC1: H[b, o] = sum_i A[b, i] * W[o, i] + bias[o]   (all-bf16 operands)
//   A[b, i] = P2bf[((i>>6)*B + b)*64 + (i&63)]  -- [V2,B,64] -> [B,4096] fused
//   M = 512, N = 512, K = 4096 ; one wave = 16x64 output (4 tiles, A reused)
//   Output H stored directly in bf16 for FC2.
// ---------------------------------------------------------------------------
__global__ void k_fc1_gemm(const unsigned short* __restrict__ P2bf,
                           const unsigned short* __restrict__ Wbf,
                           const float* __restrict__ bias,
                           unsigned short* __restrict__ Hbf) {
    int wave = blockIdx.x * (blockDim.x / 32) + (threadIdx.x / 32);
    int lane = threadIdx.x & 31;
    int mt  = wave >> 3;           // 32 m-tiles
    int ntg = wave & 7;            // 8 groups of 4 n-tiles
    int m = lane & 15;
    int h = lane >> 4;
    int brow = mt * 16 + m;

    const v8f zacc = {0.f, 0.f, 0.f, 0.f, 0.f, 0.f, 0.f, 0.f};
    v8f acc[4] = {zacc, zacc, zacc, zacc};

    for (int kk = 0; kk < 4096 / 32; ++kk) {
        v8u au;
#pragma unroll
        for (int rset = 0; rset < 2; ++rset) {
            int s = kk * 32 + 16 * h + 8 * rset;                       // i, 8-aligned
            const uint4* ap = (const uint4*)(P2bf + ((size_t)(s >> 6) * BB + brow) * 64 + (s & 63));
            uint4 av = *ap;
            au[4 * rset + 0] = av.x; au[4 * rset + 1] = av.y;
            au[4 * rset + 2] = av.z; au[4 * rset + 3] = av.w;
        }
        v16bf a = __builtin_bit_cast(v16bf, au);

#pragma unroll
        for (int t = 0; t < 4; ++t) {
            int o = (ntg * 4 + t) * 16 + m;
            const uint4* wp = (const uint4*)(Wbf + (size_t)o * 4096 + kk * 32 + 16 * h);
            __builtin_prefetch(Wbf + (size_t)o * 4096 + (kk + 4) * 32 + 16 * h, 0, 3);
            uint4 b0 = wp[0];
            uint4 b1 = wp[1];
            v8u bu;
            bu[0] = b0.x; bu[1] = b0.y; bu[2] = b0.z; bu[3] = b0.w;
            bu[4] = b1.x; bu[5] = b1.y; bu[6] = b1.z; bu[7] = b1.w;
            v16bf b = __builtin_bit_cast(v16bf, bu);
            acc[t] = __builtin_amdgcn_wmma_f32_16x16x32_bf16(
                false, a, false, b, (short)0, acc[t], false, false);
        }
    }

#pragma unroll
    for (int t = 0; t < 4; ++t) {
        int o = (ntg * 4 + t) * 16 + m;
        float bv = bias[o];
#pragma unroll
        for (int rr = 0; rr < 8; ++rr)
            Hbf[(size_t)(mt * 16 + h * 8 + rr) * 512 + o] = bf1(acc[t][rr] + bv);
    }
}

// ---------------------------------------------------------------------------
// FC2: out[b, o] = sum_i H[b, i] * W[o, i] + bias[o]
//   M = 512, N = 63 (weights zero-padded to 64), K = 512 ; f32 output
// ---------------------------------------------------------------------------
__global__ void k_fc2_gemm(const unsigned short* __restrict__ Hbf,
                           const unsigned short* __restrict__ Wbf,
                           const float* __restrict__ bias, float* __restrict__ out) {
    int wave = blockIdx.x * (blockDim.x / 32) + (threadIdx.x / 32);
    int lane = threadIdx.x & 31;
    int mt = wave >> 2;            // 4 n-tiles
    int nt = wave & 3;
    int m = lane & 15;
    int h = lane >> 4;
    int brow = mt * 16 + m;
    int o    = nt * 16 + m;

    v8f acc = {0.f, 0.f, 0.f, 0.f, 0.f, 0.f, 0.f, 0.f};

    for (int kk = 0; kk < 512 / 32; ++kk) {
        v8u au;
#pragma unroll
        for (int rset = 0; rset < 2; ++rset) {
            int s = kk * 32 + 16 * h + 8 * rset;
            uint4 av = *(const uint4*)(Hbf + (size_t)brow * 512 + s);
            au[4 * rset + 0] = av.x; au[4 * rset + 1] = av.y;
            au[4 * rset + 2] = av.z; au[4 * rset + 3] = av.w;
        }
        v16bf a = __builtin_bit_cast(v16bf, au);

        const uint4* wp = (const uint4*)(Wbf + (size_t)o * 512 + kk * 32 + 16 * h);
        uint4 b0 = wp[0];
        uint4 b1 = wp[1];
        v8u bu;
        bu[0] = b0.x; bu[1] = b0.y; bu[2] = b0.z; bu[3] = b0.w;
        bu[4] = b1.x; bu[5] = b1.y; bu[6] = b1.z; bu[7] = b1.w;
        v16bf b = __builtin_bit_cast(v16bf, bu);

        acc = __builtin_amdgcn_wmma_f32_16x16x32_bf16(
            false, a, false, b, (short)0, acc, false, false);
    }

    if (o < 63) {
        float bv = bias[o];
#pragma unroll
        for (int rr = 0; rr < 8; ++rr)
            out[(size_t)(mt * 16 + h * 8 + rr) * 63 + o] = acc[rr] + bv;
    }
}

// ---------------------------------------------------------------------------
// Host-side orchestration
// ---------------------------------------------------------------------------
static inline unsigned cdiv(size_t a, size_t b) { return (unsigned)((a + b - 1) / b); }

extern "C" void kernel_launch(void* const* d_in, const int* in_sizes, int n_in,
                              void* d_out, int out_size, void* d_ws, size_t ws_size,
                              hipStream_t stream) {
    const float* x     = (const float*)d_in[0];
    const int*   rows0 = (const int*)  d_in[1];
    const int*   cols0 = (const int*)  d_in[2];
    const float* vals0 = (const float*)d_in[3];
    const int*   rows1 = (const int*)  d_in[4];
    const int*   cols1 = (const int*)  d_in[5];
    const float* vals1 = (const float*)d_in[6];
    const float* W0 = (const float*)d_in[7];  const float* b0 = (const float*)d_in[8];
    const float* W1 = (const float*)d_in[9];  const float* b1 = (const float*)d_in[10];
    const float* W2 = (const float*)d_in[11]; const float* b2 = (const float*)d_in[12];
    const float* W3 = (const float*)d_in[13]; const float* b3 = (const float*)d_in[14];
    const float* fcW1 = (const float*)d_in[15]; const float* fcb1 = (const float*)d_in[16];
    const float* fcW2 = (const float*)d_in[17]; const float* fcb2 = (const float*)d_in[18];
    float* out = (float*)d_out;

    char* wsb = (char*)d_ws;
    constexpr size_t SZ    = (size_t)V0 * BB * 32;      // floats per slot
    constexpr size_t SLOTB = SZ * sizeof(float);        // 64 MiB slot
    float* A  = (float*)(wsb);
    float* T1 = (float*)(wsb + SLOTB);
    float* T2 = (float*)(wsb + 2 * SLOTB);
    float* O  = (float*)(wsb + 3 * SLOTB);

    const int TPB = 256;

    // ---------------- layer 0 : cheby(V0, Fin=3 -> 32), scalar-gather GEMM ----
    {
        const int n = V0 * BB * 3;                      // 1,572,864
        float* X0 = T1;
        float* t1 = T1 + 2 * 1024 * 1024;
        float* t2 = T1 + 4 * 1024 * 1024;
        k_transpose_in<<<cdiv(n, TPB), TPB, 0, stream>>>(x, X0);
        k_init_t4<<<cdiv(n / 4, TPB), TPB, 0, stream>>>(X0, t1, t2, n / 4);
        k_spmm_scatter4<<<4096, TPB, 0, stream>>>(rows0, cols0, vals0, X0, t1, 1.0f, E0, BB * 3);
        k_spmm_scatter4<<<4096, TPB, 0, stream>>>(rows0, cols0, vals0, t1, t2, 2.0f, E0, BB * 3);
        k_cheby_gemm_s<3, 32><<<V0 * BB / 16 / 8, TPB, 0, stream>>>(X0, t1, t2, W0, b0, A, V0 * BB);
    }

    // ---------------- layer 1 : cheby(V0, 32 -> 32) ----------------
    {
        const int n = V0 * BB * 32;                     // 16,777,216
        k_init_t4<<<cdiv(n / 4, TPB), TPB, 0, stream>>>(A, T1, T2, n / 4);
        k_spmm_scatter4<<<4096, TPB, 0, stream>>>(rows0, cols0, vals0, A, T1, 1.0f, E0, BB * 32);
        k_spmm_scatter4<<<4096, TPB, 0, stream>>>(rows0, cols0, vals0, T1, T2, 2.0f, E0, BB * 32);
        k_cheby_gemm_v<32, 32><<<V0 * BB / 16 / 8, TPB, 0, stream>>>(A, T1, T2, W1, b1, O, V0 * BB);
    }

    // ---------------- pool 1 : [V0,B,32] -> [V1,B,32] ----------------
    k_pool4<<<cdiv((size_t)V1 * BB * 32, TPB), TPB, 0, stream>>>(O, A, V1, 32);

    // ---------------- layer 2 : cheby(V1, 32 -> 64) ----------------
    {
        const int n = V1 * BB * 32;                     // 4,194,304
        k_init_t4<<<cdiv(n / 4, TPB), TPB, 0, stream>>>(A, T1, T2, n / 4);
        k_spmm_scatter4<<<4096, TPB, 0, stream>>>(rows1, cols1, vals1, A, T1, 1.0f, E1, BB * 32);
        k_spmm_scatter4<<<4096, TPB, 0, stream>>>(rows1, cols1, vals1, T1, T2, 2.0f, E1, BB * 32);
        k_cheby_gemm_v<32, 64><<<V1 * BB / 16 / 8, TPB, 0, stream>>>(A, T1, T2, W2, b2, O, V1 * BB);
    }

    // ---------------- layer 3 : cheby(V1, 64 -> 64) ----------------
    {
        const int n = V1 * BB * 64;                     // 8,388,608
        k_init_t4<<<cdiv(n / 4, TPB), TPB, 0, stream>>>(O, A, T1, n / 4);  // t1=A, t2=T1
        k_spmm_scatter4<<<4096, TPB, 0, stream>>>(rows1, cols1, vals1, O, A, 1.0f, E1, BB * 64);
        k_spmm_scatter4<<<4096, TPB, 0, stream>>>(rows1, cols1, vals1, A, T1, 2.0f, E1, BB * 64);
        k_cheby_gemm_v<64, 64><<<V1 * BB / 16 / 8, TPB, 0, stream>>>(O, A, T1, W3, b3, T2, V1 * BB);
    }

    // ---------------- pool 2 : [V1,B,64] -> [V2,B,64] (P2 in A) ----------------
    k_pool4<<<cdiv((size_t)V2 * BB * 64, TPB), TPB, 0, stream>>>(T2, A, V2, 64);

    // ---------------- FC head (bf16 operands, one-time conversion) ----------
    unsigned short* P2bf = (unsigned short*)(wsb + SLOTB);                  // 4 MiB
    unsigned short* W1bf = (unsigned short*)(wsb + SLOTB + (16u << 20));    // 4 MiB
    unsigned short* Hbf  = (unsigned short*)(wsb + SLOTB + (32u << 20));    // 0.5 MiB
    unsigned short* W2bf = (unsigned short*)(wsb + SLOTB + (48u << 20));    // 64 KiB

    const int nP2 = V2 * BB * 64;        // 2,097,152
    const int nW1 = 512 * 4096;          // 2,097,152
    k_cvt_bf16x8<<<cdiv(nP2 / 8, TPB), TPB, 0, stream>>>(A, P2bf, nP2 / 8);
    k_cvt_bf16x8<<<cdiv(nW1 / 8, TPB), TPB, 0, stream>>>(fcW1, W1bf, nW1 / 8);
    k_cvt_w2pad<<<cdiv(64 * 512, TPB), TPB, 0, stream>>>(fcW2, W2bf);

    // FC1: [512,4096] x [4096,512] -> Hbf ; 256 waves = 32 blocks
    k_fc1_gemm<<<32, TPB, 0, stream>>>(P2bf, W1bf, fcb1, Hbf);
    // FC2: [512,512] x [512,63] -> d_out ; 128 waves = 16 blocks
    k_fc2_gemm<<<16, TPB, 0, stream>>>(Hbf, W2bf, fcb2, out);
}
